// SoftmaxAttention_87411174408374
// MI455X (gfx1250) — matrix operand order repaired and checked by reading, not compiled, not run on hardware
//
#include <hip/hip_runtime.h>

#define HEAD_DIM 64
#define S_LEN    2048
#define NEG_INF  (-1000000000.0f)

typedef __attribute__((ext_vector_type(16))) _Float16 v16h;
typedef __attribute__((ext_vector_type(8)))  float    v8f;

// 8 waves (wave32) = 256 threads. One block per (b*h, 16-query-row tile).
// Wave w owns k-columns [w*256, w*256+255] (16 WMMA column tiles).
__global__ __launch_bounds__(256)
void SoftmaxAttention_87411174408374_kernel(
    const float* __restrict__ q, const float* __restrict__ k,
    const float* __restrict__ v, const int* __restrict__ amask,
    float* __restrict__ out, float* __restrict__ probs)
{
    __shared__ _Float16 stag[8][16][32];   // per-wave P transpose staging (8 KB)
    __shared__ float red_max[8][16];       // cross-wave row-max partials
    __shared__ float red_sum[8][16];       // cross-wave row-sum partials
    __shared__ float pout[8][16 * 64];     // per-wave PV partials (32 KB)

    const int tid  = threadIdx.x;
    const int lane = tid & 31;
    const int w    = tid >> 5;            // wave id 0..7
    const int qt   = blockIdx.x;          // q tile 0..127
    const int bh   = blockIdx.y;          // 0..23
    const int b    = bh / 12;
    const int q0   = qt * 16;

    const int nlo  = lane & 15;           // column / row-in-tile selector
    const int kgrp = lane >> 4;           // half-wave group

    const size_t base = (size_t)bh * S_LEN * HEAD_DIM;
    const float* qb = q + base;
    const float* kb = k + base;
    const float* vb = v + base;
    const int*   mb = amask + b * S_LEN;

    // ---- Q A-fragments: rows q0..q0+15, head dim K=0..63 (two K=32 frags) ----
    // A 16x32 f16 frag, contiguous k order used here: lane row = lane&15; element j -> K = kgrp*16 + j
    const float* qrow = qb + (size_t)(q0 + nlo) * HEAD_DIM;
    v16h aQ0, aQ1;
    #pragma unroll
    for (int j = 0; j < 16; ++j) {
        const int kk = kgrp * 16 + j;
        aQ0[j] = (_Float16)qrow[kk];
        aQ1[j] = (_Float16)qrow[kk + 32];
    }

    // ---- Phase 1: S = scale(QK^T) with key mask, 16 tiles kept in registers ----
    v8f   sc[16];
    float pmax[8];
    #pragma unroll
    for (int i = 0; i < 8; ++i) pmax[i] = NEG_INF;

    #pragma unroll
    for (int t = 0; t < 16; ++t) {
        const int nt   = w * 16 + t;
        const int kpos = nt * 16 + nlo;               // this lane's k column
        // B 32x16 f16 frag, contiguous k order used here: lane col = lane&15; element j -> K = kgrp*16 + j
        const float* krow = kb + (size_t)kpos * HEAD_DIM + kgrp * 16;
        v16h b0, b1;
        #pragma unroll
        for (int j = 0; j < 16; ++j) {
            b0[j] = (_Float16)krow[j];        // K head-dims 0..31
            b1[j] = (_Float16)krow[j + 32];   // K head-dims 32..63
        }
        v8f c = {};
        c = __builtin_amdgcn_wmma_f32_16x16x32_f16(false, aQ0, false, b0, (short)0, c, false, false);
        c = __builtin_amdgcn_wmma_f32_16x16x32_f16(false, aQ1, false, b1, (short)0, c, false, false);
        const bool keep = (mb[kpos] != 0);
        #pragma unroll
        for (int i = 0; i < 8; ++i) {
            const float s = keep ? c[i] * 0.125f : NEG_INF;  // 1/sqrt(64)
            sc[t][i] = s;
            pmax[i]  = fmaxf(pmax[i], s);
        }
    }

    // ---- Row max: half-wave shuffle reduce, then cross-wave via LDS ----
    #pragma unroll
    for (int i = 0; i < 8; ++i) {
        float m0 = pmax[i];
        #pragma unroll
        for (int xm = 1; xm <= 8; xm <<= 1)
            m0 = fmaxf(m0, __shfl_xor(m0, xm, 32));
        pmax[i] = m0;
    }
    if (nlo == 0) {
        #pragma unroll
        for (int i = 0; i < 8; ++i) red_max[w][kgrp * 8 + i] = pmax[i];
    }
    __syncthreads();

    float rmax[8], psum[8];
    #pragma unroll
    for (int i = 0; i < 8; ++i) {
        float m0 = NEG_INF;
        #pragma unroll
        for (int ww = 0; ww < 8; ++ww) m0 = fmaxf(m0, red_max[ww][kgrp * 8 + i]);
        rmax[i] = m0;
        psum[i] = 0.0f;
    }

    // ---- exp in place + row sums ----
    #pragma unroll
    for (int t = 0; t < 16; ++t) {
        #pragma unroll
        for (int i = 0; i < 8; ++i) {
            const float e = __expf(sc[t][i] - rmax[i]);   // masked -> underflow to 0
            sc[t][i] = e;
            psum[i] += e;
        }
    }
    #pragma unroll
    for (int i = 0; i < 8; ++i) {
        float s0 = psum[i];
        #pragma unroll
        for (int xm = 1; xm <= 8; xm <<= 1)
            s0 += __shfl_xor(s0, xm, 32);
        psum[i] = s0;
    }
    if (nlo == 0) {
        #pragma unroll
        for (int i = 0; i < 8; ++i) red_sum[w][kgrp * 8 + i] = psum[i];
    }
    __syncthreads();

    float inv[8];
    #pragma unroll
    for (int i = 0; i < 8; ++i) {
        float s0 = 0.0f;
        #pragma unroll
        for (int ww = 0; ww < 8; ++ww) s0 += red_sum[ww][kgrp * 8 + i];
        inv[i] = 1.0f / s0;
    }

    // ---- Normalize in registers + stream probs to global (bandwidth-dominant) ----
    float* pr = probs + ((size_t)bh * S_LEN + q0) * S_LEN;
    #pragma unroll
    for (int t = 0; t < 16; ++t) {
        const int col = (w * 16 + t) * 16 + nlo;
        #pragma unroll
        for (int i = 0; i < 8; ++i) {
            const float p = sc[t][i] * inv[i];
            sc[t][i] = p;
            pr[(size_t)(kgrp * 8 + i) * S_LEN + col] = p;
        }
    }

    // ---- Phase 2: O += P * V over this wave's k-span (8 chunks of K=32) ----
    v8f acc[4] = {};
    #pragma unroll
    for (int c = 0; c < 8; ++c) {
        // stage two 16x16 prob tiles as row-major 16x32 f16 (C-layout -> memory)
        #pragma unroll
        for (int tt = 0; tt < 2; ++tt) {
            const int t = c * 2 + tt;
            #pragma unroll
            for (int i = 0; i < 8; ++i)
                stag[w][kgrp * 8 + i][tt * 16 + nlo] = (_Float16)sc[t][i];
        }
        // gather back as WMMA A-fragment (same-wave DS ops are in-order)
        v16h aP;
        #pragma unroll
        for (int j = 0; j < 16; ++j) {
            const int kk = kgrp * 16 + j;
            aP[j] = stag[w][nlo][kk];
        }
        const int kstart = w * 256 + c * 32;
        #pragma unroll
        for (int dt = 0; dt < 4; ++dt) {
            // V B-fragment: col d = dt*16 + nlo, rows K = kstart + kgrp*16 + j
            const float* vcol = vb + (size_t)(kstart + kgrp * 16) * HEAD_DIM + dt * 16 + nlo;
            v16h bV;
            #pragma unroll
            for (int j = 0; j < 16; ++j)
                bV[j] = (_Float16)vcol[j * HEAD_DIM];
            acc[dt] = __builtin_amdgcn_wmma_f32_16x16x32_f16(false, aP, false, bV,
                                                             (short)0, acc[dt], false, false);
        }
    }

    // ---- deterministic cross-wave reduction of PV partials ----
    #pragma unroll
    for (int dt = 0; dt < 4; ++dt) {
        #pragma unroll
        for (int i = 0; i < 8; ++i)
            pout[w][(kgrp * 8 + i) * HEAD_DIM + dt * 16 + nlo] = acc[dt][i];
    }
    __syncthreads();

    float* ob = out + ((size_t)bh * S_LEN + q0) * HEAD_DIM;
    const int idx0 = tid * 4;   // 256 threads x 4 = 16*64 outputs
    float r[4] = {0.f, 0.f, 0.f, 0.f};
    #pragma unroll
    for (int ww = 0; ww < 8; ++ww) {
        #pragma unroll
        for (int e = 0; e < 4; ++e) r[e] += pout[ww][idx0 + e];
    }
    #pragma unroll
    for (int e = 0; e < 4; ++e) ob[idx0 + e] = r[e];
}

extern "C" void kernel_launch(void* const* d_in, const int* in_sizes, int n_in,
                              void* d_out, int out_size, void* d_ws, size_t ws_size,
                              hipStream_t stream) {
    const float* q = (const float*)d_in[0];
    const float* k = (const float*)d_in[1];
    const float* v = (const float*)d_in[2];
    const int*   m = (const int*)d_in[3];

    float* out   = (float*)d_out;
    float* probs = out + (size_t)2 * 12 * S_LEN * HEAD_DIM;  // out first, probs second

    dim3 grid(S_LEN / 16, 2 * 12);
    SoftmaxAttention_87411174408374_kernel<<<grid, 256, 0, stream>>>(q, k, v, m, out, probs);
}
